// STDP_6777458393300
// MI455X (gfx1250) — matrix operand order repaired and checked by reading, not compiled, not run on hardware
//
#include <hip/hip_runtime.h>

typedef __attribute__((ext_vector_type(16))) _Float16 v16h;
typedef __attribute__((ext_vector_type(8)))  float    v8f;

#define T_STEPS 15
#define N_BATCH 16
#define CPRE    6
#define HWPRE   64
#define CPOST   32
#define HWPOST  60
#define KSZ     5
#define P_POST  (N_BATCH * CPOST * HWPOST * HWPOST)   // 1,843,200 (multiple of 16)
#define P_PRE   (N_BATCH * CPRE  * HWPRE  * HWPRE)    //   393,216 (multiple of 16)
#define M_PER_N (CPOST * HWPOST * HWPOST)             //   115,200
#define KMAX    16

struct Winner { int c, y, x, act; float post_t; float pad0, pad1, pad2; };  // 32 B

// Order-preserving float->uint encoding so atomicMax(uint) == float max.
__device__ __forceinline__ unsigned enc_f32(float f) {
  unsigned u = __float_as_uint(f);
  return (u & 0x80000000u) ? ~u : (u | 0x80000000u);
}
__device__ __forceinline__ float dec_f32(unsigned u) {
  return (u & 0x80000000u) ? __uint_as_float(u & 0x7FFFFFFFu)
                           : __uint_as_float(~u);
}

__global__ void init_ws(unsigned* max_enc) {
  if (threadIdx.x == 0 && blockIdx.x == 0) *max_enc = enc_f32(-__builtin_inff());
}

// ---------------------------------------------------------------------------
// Temporal reduction via V_WMMA_F32_16X16X32_F16.
//   D = A(16 pos x 32 t) * B(32 t x 16), B col0 = time coeffs (T..1), col1 = 1.
//   => D col0 = weighted time sum, D col1 = spike count, per position. Exact
//   in f16 (operands are {0,1} and integers <= 15; accumulate in f32).
// A layout (wave32, 16-bit A 16x32): lanes 0-15 halves 0..7 = K 0..7;
//   lanes 16-31 halves 0..7 = K 8..15; halves 8..15 = K 16..31 (zero, T=15).
// B layout: lane&15 = column, halves = K rows (K = half + 8*(lane>=16) for halves 0..7; halves 8..15 zero).
// D layout: d[r] at M = r + 8*(lane>=16), N = lane&15.
// ---------------------------------------------------------------------------
template <bool POST>
__global__ void __launch_bounds__(256)
time_reduce_wmma(const float* __restrict__ spikes,       // (T_STEPS, P)
                 const float* __restrict__ potentials,   // (P) [POST only]
                 float* __restrict__ out_wt,             // (P) weighted sum
                 float* __restrict__ out_cnt,            // (P) count [POST only]
                 unsigned* __restrict__ max_enc,         // [POST only]
                 int P) {
  __shared__ float sred[256];
  const int lane = threadIdx.x & 31;
  const int m    = lane & 15;    // position-in-tile (A rows) / column (B,D)
  const int hi   = lane >> 4;    // K half selector
  const int wavesPerBlock = blockDim.x >> 5;
  const int waveId     = blockIdx.x * wavesPerBlock + (threadIdx.x >> 5);
  const int totalWaves = gridDim.x * wavesPerBlock;
  const int numTiles   = P >> 4;

  // B is loop-invariant: col0 = 15-k (k<15), col1 = 1 (k<15), else 0.
  v16h b = {};
  {
    if (m == 0) {
      for (int i = 0; i < 8; ++i) { const int t = hi * 8 + i; if (t < T_STEPS) b[i] = (_Float16)(float)(T_STEPS - t); }
    } else if (m == 1) {
      for (int i = 0; i < 8; ++i) { const int t = hi * 8 + i; if (t < T_STEPS) b[i] = (_Float16)1.0f; }
    }
  }

  float runMax = -__builtin_inff();

  for (int tile = waveId; tile < numTiles; tile += totalWaves) {
    const int base = tile << 4;
    const int pos  = base + m;
    v16h a = {};
    bool sawOne = false, sawZero = false;
    #pragma unroll
    for (int i = 0; i < 7; ++i) {          // t = hi*8 + i in [0..6] or [8..14]
      const int t = hi * 8 + i;
      const float s = spikes[(size_t)t * (size_t)P + (size_t)pos];
      a[i] = (_Float16)s;
      if (POST) { sawOne |= (s != 0.0f); sawZero |= (s == 0.0f); }
    }
    if (hi == 0) {                          // t = 7 (t = 15 is the zero pad)
      const float s = spikes[(size_t)7 * (size_t)P + (size_t)pos];
      a[7] = (_Float16)s;
      if (POST) { sawOne |= (s != 0.0f); sawZero |= (s == 0.0f); }
    }

    v8f acc = {};
    acc = __builtin_amdgcn_wmma_f32_16x16x32_f16(false, a, false, b,
                                                 (short)0, acc, false, false);

    if (m == 0) {                           // column 0: weighted sums
      #pragma unroll
      for (int r = 0; r < 8; ++r) out_wt[base + hi * 8 + r] = acc[r];
    }
    if (POST && m == 1) {                   // column 1: counts
      #pragma unroll
      for (int r = 0; r < 8; ++r) out_cnt[base + hi * 8 + r] = acc[r];
    }
    if (POST) {                             // max over p*s for this lane's t's
      const float p = potentials[pos];
      float cand = -__builtin_inff();
      if (sawOne)  cand = p;
      if (sawZero) cand = fmaxf(cand, 0.0f);
      runMax = fmaxf(runMax, cand);
    }
  }

  if (POST) {
    sred[threadIdx.x] = runMax;
    __syncthreads();
    for (int s = (int)blockDim.x >> 1; s > 0; s >>= 1) {
      if ((int)threadIdx.x < s)
        sred[threadIdx.x] = fmaxf(sred[threadIdx.x], sred[threadIdx.x + s]);
      __syncthreads();
    }
    if (threadIdx.x == 0) atomicMax(max_enc, enc_f32(sred[0]));
  }
}

// ---------------------------------------------------------------------------
// Per-batch iterative top-k (k sequential argmax passes with exclusion) over
// total = p*count + max_pot*post_times. Ties -> lowest index (== lax.top_k).
// ---------------------------------------------------------------------------
__global__ void __launch_bounds__(1024)
topk_kernel(const float* __restrict__ potentials,
            const float* __restrict__ post_wt,
            const float* __restrict__ post_cnt,
            const unsigned* __restrict__ max_enc,
            const int* __restrict__ kwta_p,
            Winner* __restrict__ winners) {
  __shared__ int   chosen[KMAX];
  __shared__ float svals[1024];
  __shared__ int   sidx[1024];

  const int n   = blockIdx.x;
  const int tid = threadIdx.x;
  const int nt  = blockDim.x;
  const float maxpot = dec_f32(*max_enc);
  int kwta = *kwta_p; if (kwta > KMAX) kwta = KMAX; if (kwta < 0) kwta = 0;

  const float* p_n   = potentials + (size_t)n * M_PER_N;
  const float* wt_n  = post_wt    + (size_t)n * M_PER_N;
  const float* cnt_n = post_cnt   + (size_t)n * M_PER_N;

  for (int j = 0; j < kwta; ++j) {
    float best = -__builtin_inff();
    int   bidx = 0x7FFFFFFF;
    for (int i = tid; i < M_PER_N; i += nt) {
      bool skip = false;
      for (int q = 0; q < j; ++q) skip |= (chosen[q] == i);
      if (skip) continue;
      const float tot = p_n[i] * cnt_n[i] + maxpot * wt_n[i];
      if (tot > best) { best = tot; bidx = i; }   // ascending i => lowest idx kept
    }
    svals[tid] = best; sidx[tid] = bidx;
    __syncthreads();
    for (int s = nt >> 1; s > 0; s >>= 1) {
      if (tid < s) {
        const float v2 = svals[tid + s]; const int i2 = sidx[tid + s];
        if (v2 > svals[tid] || (v2 == svals[tid] && i2 < sidx[tid])) {
          svals[tid] = v2; sidx[tid] = i2;
        }
      }
      __syncthreads();
    }
    if (tid == 0) {
      const int idx = sidx[0];
      chosen[j] = idx;
      Winner w;
      w.c = idx / (HWPOST * HWPOST);
      const int rem = idx % (HWPOST * HWPOST);
      w.y = rem / HWPOST;
      w.x = rem % HWPOST;
      w.act    = (p_n[idx] > 0.0f) ? 1 : 0;
      w.post_t = wt_n[idx];
      w.pad0 = w.pad1 = w.pad2 = 0.0f;
      winners[n * KMAX + j] = w;
    }
    __syncthreads();
  }
}

// ---------------------------------------------------------------------------
// dw[fm,cpre,ky,kx] = stab * sum_n lr_n; per (n,fm) the LAST active winner
// with c==fm wins (sequential-overwrite semantics of the reference).
// ---------------------------------------------------------------------------
__global__ void __launch_bounds__(256)
dw_kernel(const float* __restrict__ pre_times,   // (N,CPRE,64,64)
          const float* __restrict__ weight,      // (CPOST,CPRE,5,5)
          const float* __restrict__ ltp, const float* __restrict__ ltd,
          const int* __restrict__ kwta_p,
          const Winner* __restrict__ winners,
          float* __restrict__ dw) {
  const int tid = blockIdx.x * blockDim.x + threadIdx.x;
  const int TOT = CPOST * CPRE * KSZ * KSZ;      // 4800
  if (tid >= TOT) return;
  const int fm   = tid / (CPRE * KSZ * KSZ);
  const int e    = tid % (CPRE * KSZ * KSZ);
  const int cpre = e / (KSZ * KSZ);
  const int ky   = (e % (KSZ * KSZ)) / KSZ;
  const int kx   = e % KSZ;
  int kwta = *kwta_p; if (kwta > KMAX) kwta = KMAX; if (kwta < 0) kwta = 0;

  const float tp = ltp[fm], td = ltd[fm];
  float acc = 0.0f;
  for (int n = 0; n < N_BATCH; ++n) {
    int have = 0, wy = 0, wx = 0;
    float wpt = 0.0f;
    for (int j = 0; j < kwta; ++j) {
      const Winner w = winners[n * KMAX + j];
      if (w.act && w.c == fm) { have = 1; wy = w.y; wx = w.x; wpt = w.post_t; }
    }
    if (have) {
      const float pre =
          pre_times[((size_t)(n * CPRE + cpre) * HWPRE + (wy + ky)) * HWPRE + (wx + kx)];
      acc += (pre >= wpt) ? tp : td;
    }
  }
  const float w = weight[tid];
  dw[tid] = acc * (w - 0.0f) * (1.0f - w);   // stabilizer, LOWER=0, UPPER=1
}

extern "C" void kernel_launch(void* const* d_in, const int* in_sizes, int n_in,
                              void* d_out, int out_size, void* d_ws, size_t ws_size,
                              hipStream_t stream) {
  (void)in_sizes; (void)n_in; (void)out_size; (void)ws_size;
  const float* spikes_pre  = (const float*)d_in[0];
  const float* spikes_post = (const float*)d_in[1];
  const float* potentials  = (const float*)d_in[2];
  const float* weight      = (const float*)d_in[3];
  const float* ltp         = (const float*)d_in[4];
  const float* ltd         = (const float*)d_in[5];
  const int*   kwta        = (const int*)d_in[6];

  char* ws = (char*)d_ws;
  unsigned* max_enc = (unsigned*)ws;                                   // 4 B
  Winner*   winners = (Winner*)(ws + 256);                             // 8 KB
  float* pre_times  = (float*)(ws + 16384);                            // 1.5 MB
  float* post_wt    = (float*)(ws + 16384 + (size_t)P_PRE * 4);        // 7.0 MB
  float* post_cnt   = (float*)(ws + 16384 + (size_t)P_PRE * 4 + (size_t)P_POST * 4);

  init_ws<<<1, 64, 0, stream>>>(max_enc);
  time_reduce_wmma<true ><<<512, 256, 0, stream>>>(spikes_post, potentials,
                                                   post_wt, post_cnt, max_enc, P_POST);
  time_reduce_wmma<false><<<128, 256, 0, stream>>>(spikes_pre, nullptr,
                                                   pre_times, nullptr, nullptr, P_PRE);
  topk_kernel<<<N_BATCH, 1024, 0, stream>>>(potentials, post_wt, post_cnt,
                                            max_enc, kwta, winners);
  dw_kernel<<<(CPOST * CPRE * KSZ * KSZ + 255) / 256, 256, 0, stream>>>(
      pre_times, weight, ltp, ltd, kwta, winners, (float*)d_out);
}
